// GNN_graphpred_64183991271588
// MI455X (gfx1250) — matrix operand order repaired and checked by reading, not compiled, not run on hardware
//
#include <hip/hip_runtime.h>

typedef __bf16 bf16_t;
typedef __attribute__((ext_vector_type(16))) __bf16 v16bf;
typedef __attribute__((ext_vector_type(8)))  float  v8f;

#define NNODE   100000
#define EDGES   200000
#define DIM     512
#define DIM2    1024
#define NLAYER  5
#define NGRAPH  5000
#define NPG     20
#define NTASK   12
#define NUMBOND 6
#define NUMDIR  3
#define BNINV   0.9999950000374997f   /* 1/sqrt(1+1e-5) */

// ---------------------------------------------------------------------------
// fp32 -> bf16 conversion (weights once; agg each layer)
// ---------------------------------------------------------------------------
__global__ void k_cvt(const float* __restrict__ in, bf16_t* __restrict__ out, long n) {
    long i = ((long)blockIdx.x * blockDim.x + threadIdx.x) * 4;
    #pragma unroll
    for (int j = 0; j < 4; ++j)
        if (i + j < n) out[i + j] = (bf16_t)in[i + j];
}

// ---------------------------------------------------------------------------
// h = emb1[atom] + emb2[chir]  (stored bf16)
// ---------------------------------------------------------------------------
__global__ void k_embed(const int* __restrict__ x, const float* __restrict__ emb1,
                        const float* __restrict__ emb2, bf16_t* __restrict__ h) {
    int idx = blockIdx.x * blockDim.x + threadIdx.x;      // over N*D/4
    if (idx >= NNODE * (DIM / 4)) return;
    int i = idx >> 7;
    int d = (idx & 127) * 4;
    int a0 = x[i * 3 + 0], a1 = x[i * 3 + 1];
    #pragma unroll
    for (int j = 0; j < 4; ++j)
        h[(long)i * DIM + d + j] = (bf16_t)(emb1[a0 * DIM + d + j] + emb2[a1 * DIM + d + j]);
}

// ---------------------------------------------------------------------------
// agg = h + selfloop_const   (fp32 accumulator init)
// ---------------------------------------------------------------------------
__global__ void k_agg_init(const bf16_t* __restrict__ h, const float* __restrict__ e1l,
                           const float* __restrict__ e2l, float* __restrict__ agg) {
    int idx = blockIdx.x * blockDim.x + threadIdx.x;      // over N*D/4
    if (idx >= NNODE * (DIM / 4)) return;
    int i = idx >> 7;
    int d = (idx & 127) * 4;
    #pragma unroll
    for (int j = 0; j < 4; ++j)
        agg[(long)i * DIM + d + j] = (float)h[(long)i * DIM + d + j]
                                   + e1l[4 * DIM + d + j]     // SELF_LOOP_BOND = 4
                                   + e2l[0 * DIM + d + j];    // dir 0
}

// ---------------------------------------------------------------------------
// agg[dst] += h[src] + edge_emb   (fp32 atomics)
// ---------------------------------------------------------------------------
__global__ void k_edges(const bf16_t* __restrict__ h, const int* __restrict__ src,
                        const int* __restrict__ dst, const int* __restrict__ ea,
                        const float* __restrict__ e1l, const float* __restrict__ e2l,
                        float* __restrict__ agg) {
    int idx = blockIdx.x * blockDim.x + threadIdx.x;      // over E*D/4
    if (idx >= EDGES * (DIM / 4)) return;
    int e = idx >> 7;
    int d = (idx & 127) * 4;
    int s = src[e], t = dst[e];
    int bond = ea[2 * e + 0], dir = ea[2 * e + 1];
    #pragma unroll
    for (int j = 0; j < 4; ++j) {
        float v = (float)h[(long)s * DIM + d + j]
                + e1l[bond * DIM + d + j] + e2l[dir * DIM + d + j];
        atomicAdd(&agg[(long)t * DIM + d + j], v);
    }
}

// ---------------------------------------------------------------------------
// WMMA GEMM:  out[M,Nc] = epilogue(A[M,K] @ B[K,Nc]),  A,B bf16, fp32 accum
//   EPI 0 : out = bf16(relu(acc + bias))
//   EPI 1 : v = g*BNINV*(acc+bias)+beta; relu if doRelu; bf16
// block = 256 threads (8 waves); tile 64x128; K-step 32; wave tile 16x64
// A tile staged with global_load_async_to_lds_b128 (ASYNCcnt), B staged
// transposed via VGPRs; both LDS strides padded to 40 bf16 so fragment
// reads are aligned ds_load_b128 pairs.
// ---------------------------------------------------------------------------
template <int EPI>
__global__ __launch_bounds__(256)
void gemm_wmma(const bf16_t* __restrict__ A, const bf16_t* __restrict__ B,
               const float* __restrict__ bias, const float* __restrict__ gscale,
               const float* __restrict__ gshift, int doRelu,
               bf16_t* __restrict__ out, int M, int K, int Nc) {
    __shared__ alignas(16) bf16_t As[64 * 40];    // row-major,  stride 40 bf16 (80B)
    __shared__ alignas(16) bf16_t Bs[128 * 40];   // transposed [n][k], stride 40 bf16

    const int tid  = threadIdx.x;
    const int lane = tid & 31, wid = tid >> 5;
    const int wm = wid & 3, wn = wid >> 2;        // 4 M-waves x 2 N-waves
    const int kh = lane >> 4, nl = lane & 15;
    const int rowBase = blockIdx.x * 64;
    const int colBase = blockIdx.y * 128;

    v8f acc[4] = {};

    const int ar  = tid >> 2;          // 0..63 (A row)
    const int ac  = (tid & 3) * 8;     // A col chunk (8 bf16 = 16B)
    const int bk  = tid >> 3;          // 0..31 (B k-row)
    const int bn0 = (tid & 7) * 16;    // B col chunk

    // async A-tile source: clamp OOB rows to M-1 (duplicates, never stored)
    int grc = rowBase + ar;
    if (grc > M - 1) grc = M - 1;
    const bf16_t* gA = A + (long)grc * K + ac;
    const unsigned asDst =
        (unsigned)(unsigned long long)(uintptr_t)&As[ar * 40 + ac];

    for (int k0 = 0; k0 < K; k0 += 32) {
        // --- A tile 64x32 bf16: async global -> LDS (16B per thread) ---
        asm volatile("global_load_async_to_lds_b128 %0, %1, off"
                     :: "v"(asDst), "v"(gA + k0) : "memory");
        // --- B tile 32x128: stage transposed into LDS ---
        {
            const bf16_t* s = &B[(long)(k0 + bk) * Nc + colBase + bn0];
            #pragma unroll
            for (int j = 0; j < 16; ++j) Bs[(bn0 + j) * 40 + bk] = s[j];
        }
        // --- prefetch next K-step tiles into cache while WMMAs run ---
        if (k0 + 32 < K) {
            __builtin_prefetch((const void*)(gA + k0 + 32), 0, 1);
            __builtin_prefetch((const void*)&B[(long)(k0 + 32 + bk) * Nc + colBase + bn0], 0, 1);
        }
        asm volatile("s_wait_asynccnt 0x0" ::: "memory");
        __syncthreads();

        const unsigned int* As32 = (const unsigned int*)As;
        const unsigned int* Bs32 = (const unsigned int*)Bs;

        // A fragment: lane half kh selects K 0-7/16-23 vs 8-15/24-31 (ISA 16-bit A)
        union { unsigned int u[8]; v16bf v; } a;
        const int arow = wm * 16 + nl;
        #pragma unroll
        for (int v = 0; v < 8; ++v) {
            int kd = v + 4 * kh + (v >= 4 ? 4 : 0);   // dword index of K-pair
            a.u[v] = As32[arow * 20 + kd];
        }
        #pragma unroll
        for (int f = 0; f < 4; ++f) {
            // B fragment: lane half kh selects K 0-7/16-23 vs 8-15/24-31 (same as A)
            union { unsigned int u[8]; v16bf v; } b;
            const int bcol = wn * 64 + f * 16 + nl;
            #pragma unroll
            for (int v = 0; v < 8; ++v) b.u[v] = Bs32[bcol * 20 + v + 4 * kh + (v >= 4 ? 4 : 0)];
            acc[f] = __builtin_amdgcn_wmma_f32_16x16x32_bf16(
                false, a.v, false, b.v, (short)0, acc[f], false, false);
        }
        __syncthreads();
    }

    // --- epilogue: C(m,n): lanes 0-15 -> M=r, lanes 16-31 -> M=8+r; N = lane&15 ---
    #pragma unroll
    for (int f = 0; f < 4; ++f) {
        const int col = colBase + wn * 64 + f * 16 + nl;
        const float bv = bias[col];
        float gs = 0.f, gb = 0.f;
        if (EPI == 1) { gs = gscale[col] * BNINV; gb = gshift[col]; }
        #pragma unroll
        for (int r = 0; r < 8; ++r) {
            int row = rowBase + wm * 16 + kh * 8 + r;
            if (row < M) {
                float v = acc[f][r] + bv;
                if (EPI == 0) {
                    v = v > 0.f ? v : 0.f;
                } else {
                    v = gs * v + gb;
                    if (doRelu) v = v > 0.f ? v : 0.f;
                }
                out[(long)row * Nc + col] = (bf16_t)v;
            }
        }
    }
}

// ---------------------------------------------------------------------------
// head part 1: z[g,j] = BN(relu(hc . (hW1[:512,j]+hW1[512:,j]) + hb1[j]))
// (center node g*NPG; count==1 so mean==sum and pooled = [hc, hc])
// ---------------------------------------------------------------------------
__global__ __launch_bounds__(128)
void k_head1(const bf16_t* __restrict__ h, const float* __restrict__ hW1,
             const float* __restrict__ hb1, const float* __restrict__ hg,
             const float* __restrict__ hbt, float* __restrict__ z) {
    __shared__ float hc[DIM];
    int g = blockIdx.x, j = threadIdx.x;
    const bf16_t* hr = &h[(long)g * NPG * DIM];
    for (int k = j; k < DIM; k += 128) hc[k] = (float)hr[k];
    __syncthreads();
    float acc = hb1[j];
    for (int k = 0; k < DIM; ++k)
        acc += hc[k] * (hW1[k * 128 + j] + hW1[(DIM + k) * 128 + j]);
    acc = acc > 0.f ? acc : 0.f;
    z[(long)g * 128 + j] = hg[j] * acc * BNINV + hbt[j];
}

// ---------------------------------------------------------------------------
// head part 2: out[g,t] = z[g] . hW2[:,t] + hb2[t]
// ---------------------------------------------------------------------------
__global__ void k_head2(const float* __restrict__ z, const float* __restrict__ hW2,
                        const float* __restrict__ hb2, float* __restrict__ out) {
    int idx = blockIdx.x * blockDim.x + threadIdx.x;
    if (idx >= NGRAPH * NTASK) return;
    int g = idx / NTASK, t = idx - g * NTASK;
    float acc = hb2[t];
    for (int k = 0; k < 128; ++k) acc += z[(long)g * 128 + k] * hW2[k * NTASK + t];
    out[idx] = acc;
}

// ---------------------------------------------------------------------------
extern "C" void kernel_launch(void* const* d_in, const int* in_sizes, int n_in,
                              void* d_out, int out_size, void* d_ws, size_t ws_size,
                              hipStream_t stream) {
    const int*   x    = (const int*)d_in[0];
    const int*   ei   = (const int*)d_in[1];
    const int*   ea   = (const int*)d_in[2];
    const float* emb1 = (const float*)d_in[5];
    const float* emb2 = (const float*)d_in[6];
    const float* ee1  = (const float*)d_in[7];
    const float* ee2  = (const float*)d_in[8];
    const float* W1   = (const float*)d_in[9];
    const float* b1   = (const float*)d_in[10];
    const float* W2   = (const float*)d_in[11];
    const float* b2   = (const float*)d_in[12];
    const float* bng  = (const float*)d_in[13];
    const float* bnb  = (const float*)d_in[14];
    const float* hW1  = (const float*)d_in[15];
    const float* hb1  = (const float*)d_in[16];
    const float* hg   = (const float*)d_in[17];
    const float* hbt  = (const float*)d_in[18];
    const float* hW2  = (const float*)d_in[19];
    const float* hb2  = (const float*)d_in[20];

    char* ws = (char*)d_ws;
    size_t off = 0;
    auto suballoc = [&](size_t bytes) -> void* {
        void* p = ws + off;
        off = (off + bytes + 255) & ~(size_t)255;
        return p;
    };
    bf16_t* h_bf  = (bf16_t*)suballoc((size_t)NNODE * DIM * 2);      // 100 MB
    bf16_t* midbf = (bf16_t*)suballoc((size_t)NNODE * DIM2 * 2);     // 200 MB
    float*  agg   = (float*) suballoc((size_t)NNODE * DIM * 4);      // 200 MB
    bf16_t* aggbf = (bf16_t*)suballoc((size_t)NNODE * DIM * 2);      // 100 MB
    bf16_t* w1bf  = (bf16_t*)suballoc((size_t)NLAYER * DIM * DIM2 * 2);
    bf16_t* w2bf  = (bf16_t*)suballoc((size_t)NLAYER * DIM2 * DIM * 2);
    float*  zbuf  = (float*) suballoc((size_t)NGRAPH * 128 * 4);

    // weights -> bf16 (tiny: 2 x 5.25 MB)
    {
        long n = (long)NLAYER * DIM * DIM2;
        int blocks = (int)((n / 4 + 255) / 256);
        k_cvt<<<blocks, 256, 0, stream>>>(W1, w1bf, n);
        k_cvt<<<blocks, 256, 0, stream>>>(W2, w2bf, n);
    }

    k_embed<<<(NNODE * (DIM / 4) + 255) / 256, 256, 0, stream>>>(x, emb1, emb2, h_bf);

    const int* src = ei;
    const int* dst = ei + EDGES;
    const int nd4blocks = (NNODE * (DIM / 4) + 255) / 256;
    for (int l = 0; l < NLAYER; ++l) {
        const float* e1l = ee1 + (size_t)l * NUMBOND * DIM;
        const float* e2l = ee2 + (size_t)l * NUMDIR * DIM;

        k_agg_init<<<nd4blocks, 256, 0, stream>>>(h_bf, e1l, e2l, agg);
        k_edges<<<(EDGES * (DIM / 4) + 255) / 256, 256, 0, stream>>>(
            h_bf, src, dst, ea, e1l, e2l, agg);
        k_cvt<<<nd4blocks, 256, 0, stream>>>(agg, aggbf, (long)NNODE * DIM);

        dim3 g1((NNODE + 63) / 64, DIM2 / 128);
        gemm_wmma<0><<<g1, 256, 0, stream>>>(
            aggbf, w1bf + (size_t)l * DIM * DIM2, b1 + (size_t)l * DIM2,
            nullptr, nullptr, 0, midbf, NNODE, DIM, DIM2);

        dim3 g2((NNODE + 63) / 64, DIM / 128);
        gemm_wmma<1><<<g2, 256, 0, stream>>>(
            midbf, w2bf + (size_t)l * DIM2 * DIM, b2 + (size_t)l * DIM,
            bng + (size_t)l * DIM, bnb + (size_t)l * DIM,
            (l < NLAYER - 1) ? 1 : 0, h_bf, NNODE, DIM2, DIM);
    }

    k_head1<<<NGRAPH, 128, 0, stream>>>(h_bf, hW1, hb1, hg, hbt, zbuf);
    k_head2<<<(NGRAPH * NTASK + 255) / 256, 256, 0, stream>>>(zbuf, hW2, hb2, (float*)d_out);
}
